// MultiProductAttention_33947421507860
// MI455X (gfx1250) — matrix operand order repaired and checked by reading, not compiled, not run on hardware
//
#include <hip/hip_runtime.h>
#include <math.h>

// ---------------------------------------------------------------------------
// Types for CDNA5 WMMA (wave32): v_wmma_f32_16x16x32_bf16
// ---------------------------------------------------------------------------
typedef __attribute__((ext_vector_type(16))) __bf16       v16bf;
typedef __attribute__((ext_vector_type(8)))  float        v8f;
typedef __attribute__((ext_vector_type(4)))  unsigned int u32x4;
typedef __attribute__((ext_vector_type(8)))  int          i32x8;
typedef __attribute__((ext_vector_type(4)))  int          i32x4;

union Frag { u32x4 q[2]; v16bf v; };
union V8u  { u32x4 q; __bf16 e[8]; };

static __device__ __forceinline__ v8f wmma_bf16(v16bf a, v16bf b, v8f c) {
  return __builtin_amdgcn_wmma_f32_16x16x32_bf16(false, a, false, b, (short)0, c,
                                                 false, false);
}

// ---------------------------------------------------------------------------
// Tensor Data Mover: DMA a 2D bf16 tile (tileW x tileH) from global into LDS,
// with optional LDS row padding and hardware OOB zero-fill.
// ---------------------------------------------------------------------------
#if defined(__AMDGCN__) && __has_builtin(__builtin_amdgcn_tensor_load_to_lds) && \
    __has_builtin(__builtin_amdgcn_s_wait_tensorcnt)
#define HAVE_TDM 1
#else
#define HAVE_TDM 0
#endif

#if HAVE_TDM
static __device__ __forceinline__ void tdm_issue(u32x4 g0, i32x8 g1) {
  i32x4 z4 = {0, 0, 0, 0};
#if __clang_major__ >= 23
  i32x8 z8 = {0, 0, 0, 0, 0, 0, 0, 0};
  __builtin_amdgcn_tensor_load_to_lds(g0, g1, z4, z4, z8, 0);
#else
  __builtin_amdgcn_tensor_load_to_lds(g0, g1, z4, z4, 0);
#endif
}

// padAmountCode < 0 -> no padding. Codes per D# group1: interval 4 -> 32 dwords,
// 5 -> 64 dwords; amount 3 -> 4 dwords (16B).
static __device__ __forceinline__ void tdm_load_2d(const void* gptr,
                                                   unsigned ldsOff,
                                                   unsigned tileW, unsigned tileH,
                                                   unsigned dim0, unsigned dim1,
                                                   unsigned strideElems,
                                                   int padIntervalCode,
                                                   int padAmountCode) {
  unsigned long long ga = (unsigned long long)gptr;
  u32x4 g0 = {0u, 0u, 0u, 0u};
  g0[0] = 1u;                                     // count=1, no gather
  g0[1] = ldsOff;                                 // LDS byte address
  g0[2] = (unsigned)ga;                           // global_addr[31:0]
  g0[3] = (unsigned)((ga >> 32) & 0x01FFFFFFu) | (2u << 30);  // addr[56:32]|type=2
  unsigned d0 = 1u << 16;                         // data_size = 2 bytes
  if (padAmountCode >= 0)
    d0 |= (1u << 20) | ((unsigned)padIntervalCode << 22) |
          ((unsigned)padAmountCode << 25);
  i32x8 g1 = {0, 0, 0, 0, 0, 0, 0, 0};
  g1[0] = (int)d0;
  g1[1] = (int)((dim0 & 0xFFFFu) << 16);                       // tensor_dim0 lo
  g1[2] = (int)((dim0 >> 16) | ((dim1 & 0xFFFFu) << 16));      // dim0 hi|dim1 lo
  g1[3] = (int)((dim1 >> 16) | (tileW << 16));                 // dim1 hi|tile_dim0
  g1[4] = (int)tileH;                                          // tile_dim1
  g1[5] = (int)strideElems;                                    // dim0_stride lo
  tdm_issue(g0, g1);
}
#endif

// ---------------------------------------------------------------------------
// fp32 -> bf16 conversion
// ---------------------------------------------------------------------------
__global__ void f32_to_bf16_kernel(const float* __restrict__ in,
                                   __bf16* __restrict__ out, long long nElem) {
  long long i = (long long)blockIdx.x * blockDim.x + threadIdx.x;
  if (i < nElem) out[i] = (__bf16)in[i];
}

// ---------------------------------------------------------------------------
// GEMM: C[M,N] (f32) = A[M,K] (bf16, row-major) * B[N,K]^T (bf16, row-major)
// Block: 256 threads = 8 waves. Tile 256x128, K-step 64.
// Waves: 4 (M, 64 rows) x 2 (N, 64 cols); per wave 4x4 wmma per 32-wide kstep.
// Tiles arrive via tensor_load_to_lds (TDM) with pad_enable -> LDS stride 72.
// ---------------------------------------------------------------------------
#define G_BM 256
#define G_BN 128
#define G_BK 64
#define G_STR 72  // padded LDS row stride (bf16 elems)
#define G_SA_OFF 0u
#define G_SA_BYTES (G_BM * G_STR * 2)          // 36864
#define G_SB_OFF ((unsigned)G_SA_BYTES)
#define G_SB_BYTES (G_BN * G_STR * 2)          // 18432
#define G_LDS_BYTES (G_SA_BYTES + G_SB_BYTES)  // 55296

__global__ __launch_bounds__(256) void gemm_bf16_nt(const __bf16* __restrict__ A,
                                                    const __bf16* __restrict__ B,
                                                    float* __restrict__ C,
                                                    int M, int N, int K) {
  extern __shared__ char smem[];
  __bf16* sA = (__bf16*)(smem + G_SA_OFF);
  __bf16* sB = (__bf16*)(smem + G_SB_OFF);

  const int tid  = threadIdx.x;
  const int lane = tid & 31;
  const int wave = tid >> 5;
  const int l16  = lane & 15;
  const int half = (lane < 16) ? 0 : 1;
  const int m0 = blockIdx.y * G_BM;
  const int n0 = blockIdx.x * G_BN;
  const int wm = wave & 3;   // 64 rows per wave
  const int wn = wave >> 2;  // 64 cols per wave

  v8f acc[4][4];
  const v8f vzero = {0.f, 0.f, 0.f, 0.f, 0.f, 0.f, 0.f, 0.f};
#pragma unroll
  for (int mi = 0; mi < 4; ++mi)
#pragma unroll
    for (int ni = 0; ni < 4; ++ni) acc[mi][ni] = vzero;

  const int kr0 = half ? 8 : 0;
  const int krB = half ? 8 : 0;

  for (int k0 = 0; k0 < K; k0 += G_BK) {
#if HAVE_TDM
    if (tid == 0) {
      tdm_load_2d(&A[(size_t)m0 * K + k0], G_SA_OFF, G_BK, G_BM,
                  (unsigned)(K - k0), (unsigned)(M - m0), (unsigned)K, 4, 3);
      tdm_load_2d(&B[(size_t)n0 * K + k0], G_SB_OFF, G_BK, G_BN,
                  (unsigned)(K - k0), (unsigned)(N - n0), (unsigned)K, 4, 3);
      __builtin_amdgcn_s_wait_tensorcnt(0);
    }
#else
#pragma unroll
    for (int i = 0; i < 8; ++i) {  // A: 256x64 = 2048 x 16B
      int u = tid + i * 256;
      int row = u >> 3;
      int col = (u & 7) << 3;
      u32x4 da = {0u, 0u, 0u, 0u};
      if (m0 + row < M) da = *(const u32x4*)&A[(size_t)(m0 + row) * K + k0 + col];
      *(u32x4*)&sA[row * G_STR + col] = da;
    }
#pragma unroll
    for (int i = 0; i < 4; ++i) {  // B: 128x64 = 1024 x 16B
      int u = tid + i * 256;
      int row = u >> 3;
      int col = (u & 7) << 3;
      u32x4 db = {0u, 0u, 0u, 0u};
      if (n0 + row < N) db = *(const u32x4*)&B[(size_t)(n0 + row) * K + k0 + col];
      *(u32x4*)&sB[row * G_STR + col] = db;
    }
#endif
    __syncthreads();

#pragma unroll
    for (int ks = 0; ks < 2; ++ks) {
      const int kb = ks * 32;
      Frag fb[4];
#pragma unroll
      for (int ni = 0; ni < 4; ++ni) {
        const __bf16* p = &sB[(wn * 64 + ni * 16 + l16) * G_STR + kb + krB];
        fb[ni].q[0] = *(const u32x4*)&p[0];
        fb[ni].q[1] = *(const u32x4*)&p[16];
      }
#pragma unroll
      for (int mi = 0; mi < 4; ++mi) {
        Frag fa;
        const __bf16* p = &sA[(wm * 64 + mi * 16 + l16) * G_STR + kb];
        fa.q[0] = *(const u32x4*)&p[kr0];
        fa.q[1] = *(const u32x4*)&p[16 + kr0];
#pragma unroll
        for (int ni = 0; ni < 4; ++ni)
          acc[mi][ni] = wmma_bf16(fa.v, fb[ni].v, acc[mi][ni]);
      }
    }
    __syncthreads();
  }

#pragma unroll
  for (int mi = 0; mi < 4; ++mi)
#pragma unroll
    for (int ni = 0; ni < 4; ++ni)
#pragma unroll
      for (int r = 0; r < 8; ++r) {
        int row = m0 + wm * 64 + mi * 16 + r + half * 8;
        int col = n0 + wn * 64 + ni * 16 + l16;
        if (row < M && col < N) C[(size_t)row * N + col] = acc[mi][ni][r];
      }
}

// ---------------------------------------------------------------------------
// RoPE + head expansion: builds bf16 Q/K/V laid out [b*16+h][n][128].
// ---------------------------------------------------------------------------
__global__ void rope_build_kernel(const float* __restrict__ qp,
                                  const float* __restrict__ kvp,
                                  const float* __restrict__ kvhp,
                                  __bf16* __restrict__ Q, __bf16* __restrict__ K,
                                  __bf16* __restrict__ V, int n) {
  const int j   = threadIdx.x;  // pair index 0..63
  const int pos = blockIdx.x;
  const int b   = blockIdx.y;
  const size_t row = (size_t)b * n + pos;

  const float ang = (float)pos * __expf((float)j * -0.14391156831212787f);
  float sn, cs;
  __sincosf(ang, &sn, &cs);

  const float k1 = kvp[row * 256 + 2 * j];
  const float k2 = kvp[row * 256 + 2 * j + 1];
  const float v1 = kvp[row * 256 + 128 + 2 * j];
  const float v2 = kvp[row * 256 + 128 + 2 * j + 1];
  const float rk1 = k1 * cs - k2 * sn;
  const float rk2 = k1 * sn + k2 * cs;

#pragma unroll
  for (int h = 0; h < 16; ++h) {
    const float kh = kvhp[row * 32 + h];
    const float vh = kvhp[row * 32 + 16 + h];
    const size_t o = (((size_t)(b * 16 + h)) * n + pos) * 128 + 2 * j;
    K[o]     = (__bf16)(rk1 * kh);
    K[o + 1] = (__bf16)(rk2 * kh);
    V[o]     = (__bf16)(v1 * vh);
    V[o + 1] = (__bf16)(v2 * vh);
    const float q1 = qp[row * 2048 + h * 128 + 2 * j];
    const float q2 = qp[row * 2048 + h * 128 + 2 * j + 1];
    Q[o]     = (__bf16)(q1 * cs - q2 * sn);
    Q[o + 1] = (__bf16)(q1 * sn + q2 * cs);
  }
}

// ---------------------------------------------------------------------------
// Causal flash attention. Block = 128 threads (4 waves), 64 q-rows / block.
// Q,K tiles DMA'd by the TDM (padded stride 136); V manually transposed into
// LDS (d-major, stride 72) so P·V B-fragments are two ds_load_b128.
// ---------------------------------------------------------------------------
#define AT_BM 64
#define AT_BN 64
#define AT_QS 136
#define AT_KS 136
#define AT_VS 72
#define AT_PS 72
#define AT_Q_OFF 0u
#define AT_Q_BYTES (AT_BM * AT_QS * 2)  // 17408
#define AT_K_OFF ((unsigned)AT_Q_BYTES)
#define AT_K_BYTES (AT_BN * AT_KS * 2)  // 17408
#define AT_VT_OFF (AT_K_OFF + (unsigned)AT_K_BYTES)
#define AT_VT_BYTES (128 * AT_VS * 2)  // 18432
#define AT_P_OFF (AT_VT_OFF + (unsigned)AT_VT_BYTES)
#define AT_P_BYTES (AT_BM * AT_PS * 2)  // 9216
#define AT_LDS_BYTES (AT_P_OFF + AT_P_BYTES)  // 62464

__global__ __launch_bounds__(128) void attn_kernel(const __bf16* __restrict__ Q,
                                                   const __bf16* __restrict__ K,
                                                   const __bf16* __restrict__ V,
                                                   __bf16* __restrict__ O, int n) {
  extern __shared__ char smem[];
  __bf16* sQ  = (__bf16*)(smem + AT_Q_OFF);
  __bf16* sK  = (__bf16*)(smem + AT_K_OFF);
  __bf16* sVT = (__bf16*)(smem + AT_VT_OFF);  // [d][key], stride AT_VS
  __bf16* sP  = (__bf16*)(smem + AT_P_OFF);

  const int tid  = threadIdx.x;
  const int lane = tid & 31;
  const int wave = tid >> 5;
  const int l16  = lane & 15;
  const int half = (lane < 16) ? 0 : 1;
  const int bh   = blockIdx.y;  // b*16 + h
  const int b    = bh >> 4;
  const int h    = bh & 15;
  const int q0   = blockIdx.x * AT_BM;

  const __bf16* Qp = Q + (size_t)bh * n * 128;
  const __bf16* Kp = K + (size_t)bh * n * 128;
  const __bf16* Vp = V + (size_t)bh * n * 128;
  __bf16* Op = O + ((size_t)b * n) * 2048 + h * 128;

  // ---- Q tile (64 x 128) ----
#if HAVE_TDM
  if (tid == 0)
    tdm_load_2d(&Qp[(size_t)q0 * 128], AT_Q_OFF, 128, AT_BM, 128,
                (unsigned)(n - q0), 128, 5, 3);
#else
#pragma unroll
  for (int i = 0; i < 8; ++i) {
    int u = tid + i * 128;
    int row = u >> 4;
    int col = (u & 15) << 3;
    *(u32x4*)&sQ[row * AT_QS + col] =
        *(const u32x4*)&Qp[(size_t)(q0 + row) * 128 + col];
  }
#endif

  const v8f vzero = {0.f, 0.f, 0.f, 0.f, 0.f, 0.f, 0.f, 0.f};
  float rm[8], rl[8];
  v8f oacc[8];
#pragma unroll
  for (int r = 0; r < 8; ++r) { rm[r] = -INFINITY; rl[r] = 0.f; }
#pragma unroll
  for (int f = 0; f < 8; ++f) oacc[f] = vzero;

  const int kr0  = half ? 8 : 0;
  const int krB  = half ? 8 : 0;
  const int arow = wave * 16 + l16;
  const int ktiles = blockIdx.x + 1;

  for (int kt = 0; kt < ktiles; ++kt) {
    const int k0 = kt * AT_BN;
    __syncthreads();  // previous tile fully consumed
#if HAVE_TDM
    if (tid == 0)
      tdm_load_2d(&Kp[(size_t)k0 * 128], AT_K_OFF, 128, AT_BN, 128,
                  (unsigned)(n - k0), 128, 5, 3);
#else
#pragma unroll
    for (int i = 0; i < 8; ++i) {
      int u = tid + i * 128;
      int row = u >> 4;
      int col = (u & 15) << 3;
      *(u32x4*)&sK[row * AT_KS + col] =
          *(const u32x4*)&Kp[(size_t)(k0 + row) * 128 + col];
    }
#endif
    // V: manual transposing copy into sVT[d][key] (key-major thread mapping
    // -> 16B global loads, ~2-way LDS store conflicts).
#pragma unroll
    for (int i = 0; i < 8; ++i) {
      int u = tid + i * 128;
      int key = u & 63;
      int c   = (u >> 6) << 3;  // d-chunk base
      V8u dv;
      dv.q = *(const u32x4*)&Vp[(size_t)(k0 + key) * 128 + c];
#pragma unroll
      for (int jj = 0; jj < 8; ++jj) sVT[(c + jj) * AT_VS + key] = dv.e[jj];
    }
#if HAVE_TDM
    if (tid == 0) __builtin_amdgcn_s_wait_tensorcnt(0);
#endif
    __syncthreads();

    // ---- S = Q K^T : 4 d-steps x 4 key-frags ----
    v8f s[4] = {vzero, vzero, vzero, vzero};
#pragma unroll
    for (int dk = 0; dk < 4; ++dk) {
      Frag fa;
      const __bf16* ap = &sQ[arow * AT_QS + dk * 32];
      fa.q[0] = *(const u32x4*)&ap[kr0];
      fa.q[1] = *(const u32x4*)&ap[16 + kr0];
#pragma unroll
      for (int f = 0; f < 4; ++f) {
        const __bf16* bp = &sK[(f * 16 + l16) * AT_KS + dk * 32 + krB];
        Frag fb;
        fb.q[0] = *(const u32x4*)&bp[0];
        fb.q[1] = *(const u32x4*)&bp[16];
        s[f] = wmma_bf16(fa.v, fb.v, s[f]);
      }
    }

    // ---- scale + causal mask + online softmax ----
    const int rowbase = q0 + wave * 16 + half * 8;
#pragma unroll
    for (int r = 0; r < 8; ++r) {
      const int rowg = rowbase + r;
      float mx = -INFINITY;
#pragma unroll
      for (int f = 0; f < 4; ++f) {
        float v = s[f][r] * 0.08838834764831845f;
        if (k0 + f * 16 + l16 > rowg) v = -INFINITY;
        s[f][r] = v;
        mx = fmaxf(mx, v);
      }
      mx = fmaxf(mx, __shfl_xor(mx, 1, 32));
      mx = fmaxf(mx, __shfl_xor(mx, 2, 32));
      mx = fmaxf(mx, __shfl_xor(mx, 4, 32));
      mx = fmaxf(mx, __shfl_xor(mx, 8, 32));
      const float nm   = fmaxf(rm[r], mx);
      const float corr = __expf(rm[r] - nm);
      float ls = 0.f;
#pragma unroll
      for (int f = 0; f < 4; ++f) {
        float p = __expf(s[f][r] - nm);
        s[f][r] = p;
        ls += p;
      }
      ls += __shfl_xor(ls, 1, 32);
      ls += __shfl_xor(ls, 2, 32);
      ls += __shfl_xor(ls, 4, 32);
      ls += __shfl_xor(ls, 8, 32);
      rl[r] = rl[r] * corr + ls;
      rm[r] = nm;
#pragma unroll
      for (int f = 0; f < 8; ++f) oacc[f][r] *= corr;
    }

    // ---- stage P (bf16) into LDS (per-wave private rows) ----
#pragma unroll
    for (int f = 0; f < 4; ++f)
#pragma unroll
      for (int r = 0; r < 8; ++r)
        sP[(wave * 16 + half * 8 + r) * AT_PS + f * 16 + l16] = (__bf16)s[f][r];

    // ---- O += P @ V : 2 key-steps x 8 d-frags (B from transposed sVT) ----
#pragma unroll
    for (int kk = 0; kk < 2; ++kk) {
      Frag fa;
      const __bf16* ap = &sP[arow * AT_PS + kk * 32];
      fa.q[0] = *(const u32x4*)&ap[kr0];
      fa.q[1] = *(const u32x4*)&ap[16 + kr0];
#pragma unroll
      for (int f = 0; f < 8; ++f) {
        const __bf16* bp = &sVT[(f * 16 + l16) * AT_VS + kk * 32 + krB];
        Frag fb;
        fb.q[0] = *(const u32x4*)&bp[0];
        fb.q[1] = *(const u32x4*)&bp[16];
        oacc[f] = wmma_bf16(fa.v, fb.v, oacc[f]);
      }
    }
  }

  // ---- epilogue: O / l -> bf16 [b][n][h*128 + d] ----
#pragma unroll
  for (int r = 0; r < 8; ++r) {
    const int rowg = q0 + wave * 16 + half * 8 + r;
    const float inv = 1.0f / rl[r];
#pragma unroll
    for (int f = 0; f < 8; ++f)
      Op[(size_t)rowg * 2048 + f * 16 + l16] = (__bf16)(oacc[f][r] * inv);
  }
}

// ---------------------------------------------------------------------------
// Host-side orchestration
// ---------------------------------------------------------------------------
extern "C" void kernel_launch(void* const* d_in, const int* in_sizes, int n_in,
                              void* d_out, int out_size, void* d_ws,
                              size_t ws_size, hipStream_t stream) {
  (void)in_sizes; (void)n_in; (void)out_size; (void)ws_size;

  const float* x    = (const float*)d_in[0];
  const float* Wq   = (const float*)d_in[1];
  const float* Wkv  = (const float*)d_in[2];
  const float* Wkvh = (const float*)d_in[3];
  const float* Wo   = (const float*)d_in[4];
  float* out = (float*)d_out;

  const int B = 2, N = 2048, E = 2048, H = 16, D = 128;
  const size_t nTok = (size_t)B * N;  // 4096

  char* ws = (char*)d_ws;
  size_t off = 0;
  auto take = [&](size_t bytes) -> char* {
    char* p = ws + off;
    off = (off + bytes + 255) & ~(size_t)255;
    return p;
  };

  __bf16* xb    = (__bf16*)take(nTok * E * 2);
  __bf16* Wqb   = (__bf16*)take((size_t)(H * D) * E * 2);
  __bf16* Wkvb  = (__bf16*)take((size_t)(2 * D) * E * 2);
  __bf16* Wkvhb = (__bf16*)take((size_t)(2 * H) * E * 2);
  __bf16* Wob   = (__bf16*)take((size_t)E * (H * D) * 2);
  float*  qproj = (float*)take(nTok * (size_t)(H * D) * 4);
  float*  kvprj = (float*)take(nTok * (size_t)(2 * D) * 4);
  float*  kvhp  = (float*)take(nTok * (size_t)(2 * H) * 4);
  __bf16* Qb    = (__bf16*)take((size_t)B * H * N * D * 2);
  __bf16* Kb    = (__bf16*)take((size_t)B * H * N * D * 2);
  __bf16* Vb    = (__bf16*)take((size_t)B * H * N * D * 2);
  __bf16* Ob    = (__bf16*)take(nTok * (size_t)(H * D) * 2);

  auto conv = [&](const float* src, __bf16* dst, long long ne) {
    int blocks = (int)((ne + 255) / 256);
    f32_to_bf16_kernel<<<dim3(blocks), dim3(256), 0, stream>>>(src, dst, ne);
  };
  conv(x, xb, (long long)nTok * E);
  conv(Wq, Wqb, (long long)(H * D) * E);
  conv(Wkv, Wkvb, (long long)(2 * D) * E);
  conv(Wkvh, Wkvhb, (long long)(2 * H) * E);
  conv(Wo, Wob, (long long)E * (H * D));

  auto gemm = [&](const __bf16* A, const __bf16* Bm, float* C, int M, int Nn,
                  int K) {
    dim3 grid((Nn + G_BN - 1) / G_BN, (M + G_BM - 1) / G_BM);
    gemm_bf16_nt<<<grid, 256, G_LDS_BYTES, stream>>>(A, Bm, C, M, Nn, K);
  };

  // Projections: y = x @ W^T
  gemm(xb, Wqb, qproj, (int)nTok, H * D, E);
  gemm(xb, Wkvb, kvprj, (int)nTok, 2 * D, E);
  gemm(xb, Wkvhb, kvhp, (int)nTok, 2 * H, E);

  // RoPE + head expansion -> bf16 Q/K/V per (b,h)
  rope_build_kernel<<<dim3(N, B), 64, 0, stream>>>(qproj, kvprj, kvhp, Qb, Kb,
                                                   Vb, N);

  // Causal flash attention -> Ob bf16 [b][n][h*d]
  attn_kernel<<<dim3(N / AT_BM, B * H), 128, AT_LDS_BYTES, stream>>>(Qb, Kb, Vb,
                                                                     Ob, N);

  // Output projection: out = o @ Wo^T
  gemm(Ob, Wob, out, (int)nTok, E, H * D);
}
